// hetero_genreformer_18013092839765
// MI455X (gfx1250) — matrix operand order repaired and checked by reading, not compiled, not run on hardware
//
// MI455X (gfx1250, CDNA5) implementation of the hetero GNN + genre transformer.
// All dense math (lin1/lin2, GAT projections, qkv/o/ffn, genre-concat linears)
// goes through v_wmma_f32_16x16x32_bf16 (f32 data converted to bf16 operands,
// f32 accumulation). GEMM uses a 128x128x32 block tile, 8 wave32 waves
// (64x32 per wave = 8 WMMA tiles), double-buffered LDS with a one-barrier
// software pipeline, and all staging-path branches are uniform per K-tile.
// Edge softmax / scatter and genre pooling are bandwidth-bound atomic kernels.
// Workspace requirement ~1.4 GB.

#include <hip/hip_runtime.h>
#include <cstddef>
#include <utility>

typedef __bf16 bf16_t;
typedef __attribute__((ext_vector_type(16))) __bf16 bf16x16;
typedef __attribute__((ext_vector_type(8)))  __bf16 bf16x8;
typedef __attribute__((ext_vector_type(8)))  float  f32x8;

#define HG_D 1024
#define HG_G 30

// ---------------------------------------------------------------------------
// Staging helpers (always inlined so LDS address space is preserved).
// A tile: 128 rows x 32 K (4 float4 chunks / thread).
// W tile: 32 K x 128 cols, stored transposed Bs[col][k] (4 float4 chunks).
// ---------------------------------------------------------------------------
__device__ __forceinline__ void hg_load_stage(
    const float* __restrict__ A, const float* __restrict__ A2,
    const int* __restrict__ gidx, const float* __restrict__ W,
    const float* __restrict__ pb1, const float* __restrict__ pb2,
    int preMode, int N, int K, int M, int K1,
    int rowBase, int colBase, int k0, int t,
    float4 av[4], float4 wv4[4])
{
    const bool useA2 = (k0 >= K1);        // uniform: K1, k0 multiples of 32
    const int  kk = (t & 7) * 4;
    const int  kg = k0 + kk;

    float4 pbv = make_float4(0.f, 0.f, 0.f, 0.f);
    if (preMode) {                        // uniform branch
        if (pb1) pbv = *reinterpret_cast<const float4*>(pb1 + kg);
        if (pb2) {
            float4 q = *reinterpret_cast<const float4*>(pb2 + kg);
            pbv.x += q.x; pbv.y += q.y; pbv.z += q.z; pbv.w += q.w;
        }
    }
#pragma unroll
    for (int i = 0; i < 4; ++i) {
        int r  = i * 32 + (t >> 3);       // 0..127
        int gr = rowBase + r;
        float4 v = make_float4(0.f, 0.f, 0.f, 0.f);
        if (gr < N) {
            const float* src;
            if (!useA2) src = A  + (size_t)gr * K1 + kg;
            else        src = A2 + (size_t)gidx[gr] * (size_t)(K - K1) + (kg - K1);
            v = *reinterpret_cast<const float4*>(src);
        }
        if (preMode) { v.x += pbv.x; v.y += pbv.y; v.z += pbv.z; v.w += pbv.w; }
        av[i] = v;
    }
#pragma unroll
    for (int i = 0; i < 4; ++i) {
        int kr = i * 8 + (t >> 5);        // 0..31
        int cc = (t & 31) * 4;            // 0..124
        wv4[i] = *reinterpret_cast<const float4*>(
            W + (size_t)(k0 + kr) * M + colBase + cc);
    }
}

__device__ __forceinline__ void hg_store_stage(
    bf16_t (* __restrict__ Asb)[40], bf16_t (* __restrict__ Bsb)[40],
    int t, const float4 av[4], const float4 wv4[4],
    float preScale, int preMode)
{
    const int kk = (t & 7) * 4;
#pragma unroll
    for (int i = 0; i < 4; ++i) {
        int r = i * 32 + (t >> 3);
        float xs[4] = {av[i].x, av[i].y, av[i].z, av[i].w};
#pragma unroll
        for (int j = 0; j < 4; ++j) {
            float x = xs[j];
            if (preMode) {                // uniform: bias already added at load
                x *= preScale;
                x = (x > 0.f) ? x : 0.01f * x;   // leaky 0.01 pre-activation
            }
            Asb[r][kk + j] = (bf16_t)x;
        }
    }
#pragma unroll
    for (int i = 0; i < 4; ++i) {
        int kr = i * 8 + (t >> 5);
        int cc = (t & 31) * 4;
        Bsb[cc + 0][kr] = (bf16_t)wv4[i].x;
        Bsb[cc + 1][kr] = (bf16_t)wv4[i].y;
        Bsb[cc + 2][kr] = (bf16_t)wv4[i].z;
        Bsb[cc + 3][kr] = (bf16_t)wv4[i].w;
    }
}

// ---------------------------------------------------------------------------
// Generic WMMA GEMM: Out[N,M] = post( pre(A)[N,K] @ W[K,M] + bias )
//   pre(x)  = leaky0.01( preScale * (x + pb1[k] + pb2[k]) )   when preMode=1
//   post(v) = optLeaky(postSlope)( v + bias[m] )
//   Gather-concat A: col k < K1 from A[row], else from A2[gidx[row]][k-K1].
// Block 256 threads = 8 wave32 waves; 128x128 tile; K-step 32; double-buffered
// LDS, one barrier per K-step.  Requires M % 128 == 0, K % 32 == 0.
// ---------------------------------------------------------------------------
__global__ __launch_bounds__(256)
void hg_gemm_bf16(const float* __restrict__ A,
                  const float* __restrict__ A2,
                  const int*   __restrict__ gidx,
                  const float* __restrict__ W,
                  const float* __restrict__ bias,
                  const float* __restrict__ pb1,
                  const float* __restrict__ pb2,
                  float preScale, int preMode,
                  int postAct, float postSlope,
                  float* __restrict__ Out,
                  int N, int K, int M, int K1)
{
    __shared__ __attribute__((aligned(16))) bf16_t As[2][128][40]; // stride 40 -> 80B rows, 16B aligned
    __shared__ __attribute__((aligned(16))) bf16_t Bs[2][128][40]; // transposed: Bs[buf][col][k]

    const int t       = threadIdx.x;
    const int rowBase = blockIdx.y * 128;
    const int colBase = blockIdx.x * 128;

    const int lane = t & 31;
    const int wv   = t >> 5;          // 0..7
    const int vr   = (wv >> 2) * 64;  // wave row offset: 0 / 64
    const int vc   = (wv & 3)  * 32;  // wave col offset: 0..96
    const int m16  = lane & 15;
    const int half = lane >> 4;       // 0/1

    f32x8 acc[4][2];
#pragma unroll
    for (int s = 0; s < 4; ++s)
#pragma unroll
        for (int u = 0; u < 2; ++u)
#pragma unroll
            for (int e = 0; e < 8; ++e) acc[s][u][e] = 0.0f;

    const int nk = K >> 5;

    // ---- prologue: stage tile 0 into buffer 0 ----
    {
        float4 av[4], wv4[4];
        hg_load_stage(A, A2, gidx, W, pb1, pb2, preMode, N, K, M, K1,
                      rowBase, colBase, 0, t, av, wv4);
        hg_store_stage(As[0], Bs[0], t, av, wv4, preScale, preMode);
    }
    __syncthreads();

    for (int kt = 0; kt < nk; ++kt) {
        const int cur = kt & 1;
        const bool havenext = (kt + 1 < nk);

        // issue next tile's global loads early (latency hidden behind WMMA)
        float4 av[4], wv4[4];
        if (havenext)
            hg_load_stage(A, A2, gidx, W, pb1, pb2, preMode, N, K, M, K1,
                          rowBase, colBase, (kt + 1) << 5, t, av, wv4);
        if (havenext && (kt + 2 < nk))   // gfx1250 global_prefetch two tiles ahead
            __builtin_prefetch(W + (size_t)(((kt + 2) << 5) + (t >> 5)) * M + colBase, 0, 1);

        // ---- fragments per ISA VGPR layouts ----
        bf16x16 afrag[4], bfrag[2];
#pragma unroll
        for (int s = 0; s < 4; ++s) {
            // A 16x32 bf16: lane m = lane&15; elems 0..7 -> K=kbase..+7, 8..15 -> 16+kbase..
            const bf16_t* ap = &As[cur][vr + s * 16 + m16][0];
            bf16x8 lo = *reinterpret_cast<const bf16x8*>(ap + half * 8);
            bf16x8 hi = *reinterpret_cast<const bf16x8*>(ap + 16 + half * 8);
            afrag[s] = __builtin_shufflevector(lo, hi, 0,1,2,3,4,5,6,7,8,9,10,11,12,13,14,15);
        }
#pragma unroll
        for (int u = 0; u < 2; ++u) {
            // B 32x16 bf16: lane n = lane&15; elems 0..7 -> K=half*8..+7, 8..15 -> 16+half*8..
            const bf16_t* bp = &Bs[cur][vc + u * 16 + m16][0];
            bf16x8 blo = *reinterpret_cast<const bf16x8*>(bp + half * 8);
            bf16x8 bhi = *reinterpret_cast<const bf16x8*>(bp + 16 + half * 8);
            bfrag[u] = __builtin_shufflevector(blo, bhi, 0,1,2,3,4,5,6,7,8,9,10,11,12,13,14,15);
        }
#pragma unroll
        for (int s = 0; s < 4; ++s)
#pragma unroll
            for (int u = 0; u < 2; ++u)
                acc[s][u] = __builtin_amdgcn_wmma_f32_16x16x32_bf16(
                    false, afrag[s], false, bfrag[u], (short)0, acc[s][u], false, false);

        if (havenext)
            hg_store_stage(As[cur ^ 1], Bs[cur ^ 1], t, av, wv4, preScale, preMode);
        __syncthreads();
    }

    // ---- epilogue: C/D layout row = j + 8*half, col = lane&15 ----
#pragma unroll
    for (int s = 0; s < 4; ++s)
#pragma unroll
        for (int u = 0; u < 2; ++u) {
            int c = colBase + vc + u * 16 + m16;
            float bval = bias ? bias[c] : 0.f;   // uniform null check
#pragma unroll
            for (int j = 0; j < 8; ++j) {
                int r = rowBase + vr + s * 16 + j + (half ? 8 : 0);
                if (r < N) {
                    float val = acc[s][u][j] + bval;
                    if (postAct) val = (val > 0.f) ? val : postSlope * val;
                    Out[(size_t)r * M + c] = val;
                }
            }
        }
}

// ---------------------------------------------------------------------------
// Small VALU kernels
// ---------------------------------------------------------------------------
__global__ void hg_fill(float* __restrict__ p, float v, long long n) {
    long long i = (long long)blockIdx.x * blockDim.x + threadIdx.x;
    if (i < n) p[i] = v;
}

// a[n,h] = sum_c h[n, h*256+c] * av[h*256+c]
__global__ __launch_bounds__(256)
void hg_attvec(const float* __restrict__ hm, const float* __restrict__ av,
               float* __restrict__ out, int n)
{
    int row = blockIdx.x, t = threadIdx.x;
    __shared__ float red[256];
    const float* hp = hm + (size_t)row * HG_D + t * 4;
    const float* ap = av + t * 4;
    red[t] = hp[0]*ap[0] + hp[1]*ap[1] + hp[2]*ap[2] + hp[3]*ap[3];
    __syncthreads();
    for (int o = 32; o > 0; o >>= 1) {
        if ((t & 63) < o) red[t] += red[t + o];
        __syncthreads();
    }
    if ((t & 63) == 0) out[(size_t)row * 4 + (t >> 6)] = red[t];
}

__device__ inline void hg_atomicMaxF(float* addr, float v) {
    unsigned* ua = (unsigned*)addr;
    unsigned old = *ua;
    while (__uint_as_float(old) < v) {
        unsigned assumed = old;
        old = atomicCAS(ua, assumed, __float_as_uint(v));
        if (old == assumed) break;
    }
}

__global__ void hg_edge_logits(const int* __restrict__ src, const int* __restrict__ dst,
                               const float* __restrict__ aS, const float* __restrict__ aD,
                               float* __restrict__ ebuf, float* __restrict__ m, int E)
{
    int idx = blockIdx.x * blockDim.x + threadIdx.x;
    if (idx >= E * 4) return;
    int e = idx >> 2, h = idx & 3;
    float v = aS[(size_t)src[e] * 4 + h] + aD[(size_t)dst[e] * 4 + h];
    v = (v > 0.f) ? v : 0.2f * v;            // GAT leaky slope 0.2
    ebuf[idx] = v;
    hg_atomicMaxF(&m[(size_t)dst[e] * 4 + h], v);
}

__global__ void hg_fix_m(float* __restrict__ m, int n) {
    int i = blockIdx.x * blockDim.x + threadIdx.x;
    if (i < n) { float v = m[i]; if (!(v > -3.0e38f)) m[i] = 0.f; }
}

__global__ void hg_edge_exp(const int* __restrict__ dst, float* __restrict__ ebuf,
                            const float* __restrict__ m, float* __restrict__ den, int E)
{
    int idx = blockIdx.x * blockDim.x + threadIdx.x;
    if (idx >= E * 4) return;
    int e = idx >> 2, h = idx & 3;
    float ex = __expf(ebuf[idx] - m[(size_t)dst[e] * 4 + h]);
    ebuf[idx] = ex;
    atomicAdd(&den[(size_t)dst[e] * 4 + h], ex);
}

__global__ __launch_bounds__(256)
void hg_edge_scatter(const int* __restrict__ src, const int* __restrict__ dst,
                     const float* __restrict__ hsrc, const float* __restrict__ ebuf,
                     const float* __restrict__ den, float* __restrict__ agg)
{
    int e = blockIdx.x, t = threadIdx.x;
    int s = src[e], d = dst[e];
#pragma unroll
    for (int i = 0; i < 4; ++i) {                       // head i owns channels i*256..
        int ch = i * 256 + t;
        float alpha = ebuf[(size_t)e * 4 + i] / (den[(size_t)d * 4 + i] + 1e-16f);
        atomicAdd(&agg[(size_t)d * HG_D + ch], hsrc[(size_t)s * HG_D + ch] * alpha);
    }
}

__global__ __launch_bounds__(256)
void hg_genre_scatter(const float* __restrict__ x, const int* __restrict__ gen,
                      float* __restrict__ seg, float* __restrict__ cnt)
{
    int n = blockIdx.x, t = threadIdx.x;
    int g = gen[n];
#pragma unroll
    for (int i = 0; i < 4; ++i) {
        int ch = i * 256 + t;
        atomicAdd(&seg[(size_t)g * HG_D + ch], x[(size_t)n * HG_D + ch]);
    }
    if (t == 0) atomicAdd(&cnt[g], 1.0f);
}

__global__ void hg_genre_mean(const float* __restrict__ seg, const float* __restrict__ cnt,
                              float* __restrict__ g)
{
    int i = blockIdx.x * blockDim.x + threadIdx.x;
    if (i < HG_G * HG_D) g[i] = seg[i] / fmaxf(cnt[i >> 10], 1.0f);
}

// 4-head attention on [30,1024]; qkv packed [30,3072]; key mask where cnt[g]==0
__global__ __launch_bounds__(256)
void hg_mha(const float* __restrict__ qkv, const float* __restrict__ cnt,
            float* __restrict__ out, int T)
{
    int h = blockIdx.x, t = threadIdx.x;
    __shared__ float S[HG_G][HG_G];
    __shared__ float Vb[HG_G][256];
    for (int p = t; p < T * T; p += 256) {
        int q = p / T, k = p % T;
        const float* qp = qkv + (size_t)q * 3072 + h * 256;
        const float* kp = qkv + (size_t)k * 3072 + 1024 + h * 256;
        float d = 0.f;
        for (int c = 0; c < 256; ++c) d += qp[c] * kp[c];
        d *= 0.0625f;                       // 1/sqrt(256)
        if (cnt[k] == 0.0f) d = -1e9f;
        S[q][k] = d;
    }
    for (int p = t; p < T * 256; p += 256) {
        int k = p >> 8, c = p & 255;
        Vb[k][c] = qkv[(size_t)k * 3072 + 2048 + h * 256 + c];
    }
    __syncthreads();
    if (t < T) {
        float mx = -3.4e38f;
        for (int k = 0; k < T; ++k) mx = fmaxf(mx, S[t][k]);
        float sm = 0.f;
        for (int k = 0; k < T; ++k) { float e = __expf(S[t][k] - mx); S[t][k] = e; sm += e; }
        float inv = 1.0f / sm;
        for (int k = 0; k < T; ++k) S[t][k] *= inv;
    }
    __syncthreads();
    for (int q = 0; q < T; ++q) {
        float acc = 0.f;
        for (int k = 0; k < T; ++k) acc += S[q][k] * Vb[k][t];
        out[(size_t)q * HG_D + h * 256 + t] = acc;
    }
}

// out = LayerNorm(a + b) * s + bi   (D=1024, one block per row)
__global__ __launch_bounds__(256)
void hg_ln(const float* __restrict__ a, const float* __restrict__ b,
           const float* __restrict__ s, const float* __restrict__ bi,
           float* __restrict__ out, int D)
{
    int row = blockIdx.x, t = threadIdx.x;
    __shared__ float red[256];
    float acc = 0.f;
    for (int i = t; i < D; i += 256) acc += a[(size_t)row * D + i] + b[(size_t)row * D + i];
    red[t] = acc; __syncthreads();
    for (int o = 128; o > 0; o >>= 1) { if (t < o) red[t] += red[t + o]; __syncthreads(); }
    float mu = red[0] / D;
    __syncthreads();
    float v = 0.f;
    for (int i = t; i < D; i += 256) {
        float d = a[(size_t)row * D + i] + b[(size_t)row * D + i] - mu;
        v += d * d;
    }
    red[t] = v; __syncthreads();
    for (int o = 128; o > 0; o >>= 1) { if (t < o) red[t] += red[t + o]; __syncthreads(); }
    float inv = rsqrtf(red[0] / D + 1e-5f);
    for (int i = t; i < D; i += 256) {
        float x = a[(size_t)row * D + i] + b[(size_t)row * D + i];
        out[(size_t)row * D + i] = (x - mu) * inv * s[i] + bi[i];
    }
}

// ---------------------------------------------------------------------------
// Host orchestration
// ---------------------------------------------------------------------------
struct HgLin { const float *W, *b; };
struct HgGat { const float *W, *as, *ad, *b; };
struct HgTrm { const float *Wqkv,*bqkv,*Wo,*bo,*ln1s,*ln1b,*W1,*b1,*W2,*b2,*ln2s,*ln2b; };

extern "C" void kernel_launch(void* const* d_in, const int* in_sizes, int n_in,
                              void* d_out, int out_size, void* d_ws, size_t ws_size,
                              hipStream_t stream)
{
    if (n_in < 115) return;  // expected: 9 data inputs + 106 param leaves
    (void)out_size; (void)ws_size;

    const float* x_user   = (const float*)d_in[0];
    const float* x_movie  = (const float*)d_in[1];
    const float* x_review = (const float*)d_in[2];
    const int* ei[4]   = {(const int*)d_in[3], (const int*)d_in[4],
                          (const int*)d_in[5], (const int*)d_in[6]};  // cbr,post,pby,sim
    int Ecnt[4]        = {in_sizes[3]/2, in_sizes[4]/2, in_sizes[5]/2, in_sizes[6]/2};
    const int* review_genre = (const int*)d_in[7];
    const int* movie_genre  = (const int*)d_in[8];

    const int nU = in_sizes[0] / HG_D, nM = in_sizes[1] / HG_D, nR = in_sizes[2] / HG_D;
    const int nMax = (nR > nU ? (nR > nM ? nR : nM) : (nU > nM ? nU : nM));
    int Emax = Ecnt[0];
    for (int i = 1; i < 4; ++i) if (Ecnt[i] > Emax) Emax = Ecnt[i];

    // --- param leaves: depth-first Python-dict insertion order ---
    int p = 9;
    auto F = [&](int i) { return (const float*)d_in[i]; };
    HgLin lin2[2][3]; HgGat gat[2][4]; HgLin lin1[3]; HgTrm trm[2][2]; HgLin rlin[2], mlin[2];
    for (int L = 0; L < 2; ++L) {
        for (int t = 0; t < 3; ++t) { lin2[L][t] = {F(p), F(p+1)}; p += 2; }          // user,movie,review
        for (int e = 0; e < 4; ++e) { gat[L][e] = {F(p),F(p+1),F(p+2),F(p+3)}; p += 4; } // cbr,post,pby,sim
        if (L == 0) for (int t = 0; t < 3; ++t) { lin1[t] = {F(p), F(p+1)}; p += 2; }
    }
    for (int L = 0; L < 2; ++L)
        for (int l = 0; l < 2; ++l) {
            trm[L][l] = {F(p),F(p+1),F(p+2),F(p+3),F(p+4),F(p+5),
                         F(p+6),F(p+7),F(p+8),F(p+9),F(p+10),F(p+11)};
            p += 12;
        }
    for (int L = 0; L < 2; ++L) { rlin[L] = {F(p), F(p+1)}; p += 2; }
    for (int L = 0; L < 2; ++L) { mlin[L] = {F(p), F(p+1)}; p += 2; }

    // --- workspace bump allocator (~1.4 GB required) ---
    size_t off = 0;
    auto alloc = [&](size_t nf) -> float* {
        float* r = (float*)((char*)d_ws + off);
        off += ((nf * sizeof(float) + 255) & ~(size_t)255);
        return r;
    };
    float* xgU = alloc((size_t)nU*HG_D); float* xgM = alloc((size_t)nM*HG_D); float* xgR = alloc((size_t)nR*HG_D);
    float* bufU[2] = {alloc((size_t)nU*HG_D), alloc((size_t)nU*HG_D)};
    float* bufM[2] = {alloc((size_t)nM*HG_D), alloc((size_t)nM*HG_D)};
    float* bufR[2] = {alloc((size_t)nR*HG_D), alloc((size_t)nR*HG_D)};
    float* hA   = alloc((size_t)nMax*HG_D);
    float* hB   = alloc((size_t)nMax*HG_D);
    float* aggU = alloc((size_t)nU*HG_D);
    float* aggM = alloc((size_t)nM*HG_D);
    float* aggR = alloc((size_t)nR*HG_D);
    float* aS   = alloc((size_t)nMax*4); float* aD = alloc((size_t)nMax*4);
    float* mseg = alloc((size_t)nMax*4); float* den = alloc((size_t)nMax*4);
    float* ebuf = alloc((size_t)Emax*4);
    float* seg  = alloc(HG_G*HG_D); float* cntb = alloc(64);
    float* gA   = alloc(HG_G*HG_D); float* gB = alloc(HG_G*HG_D);
    float* tqkv = alloc((size_t)HG_G*3*HG_D); float* tattn = alloc(HG_G*HG_D);
    float* ttmp = alloc(HG_G*HG_D); float* tx1 = alloc(HG_G*HG_D);
    float* tffn = alloc((size_t)HG_G*2048);

    auto fill = [&](float* ptr, float v, long long n) {
        hg_fill<<<dim3((unsigned)((n + 255) / 256)), 256, 0, stream>>>(ptr, v, n);
    };
    auto gemm = [&](const float* A, const float* A2, const int* gidx,
                    const float* W, const float* bias,
                    const float* pb1, const float* pb2, float preScale, int preMode,
                    int postAct, float postSlope,
                    float* Out, int N, int K, int M, int K1) {
        dim3 g((M + 127) / 128, (N + 127) / 128);
        hg_gemm_bf16<<<g, 256, 0, stream>>>(A, A2, gidx, W, bias, pb1, pb2,
                                            preScale, preMode, postAct, postSlope,
                                            Out, N, K, M, K1);
    };
    auto gatconv = [&](const float* xs, int ns, const float* xdd, int nd,
                       const int* e_idx, int E, const HgGat& gp,
                       float* agg, bool zero_agg, bool same) {
        gemm(xs, nullptr, nullptr, gp.W, nullptr, nullptr, nullptr, 1.f, 0, 0, 0.f,
             hA, ns, HG_D, HG_D, HG_D);                                    // h_src
        const float* hd = hA;
        if (!same) {
            gemm(xdd, nullptr, nullptr, gp.W, nullptr, nullptr, nullptr, 1.f, 0, 0, 0.f,
                 hB, nd, HG_D, HG_D, HG_D);                                // h_dst
            hd = hB;
        }
        hg_attvec<<<ns, 256, 0, stream>>>(hA, gp.as, aS, ns);
        hg_attvec<<<nd, 256, 0, stream>>>(hd, gp.ad, aD, nd);
        fill(mseg, -__builtin_inff(), (long long)nd * 4);
        fill(den, 0.f, (long long)nd * 4);
        if (zero_agg) fill(agg, 0.f, (long long)nd * HG_D);
        const int* src = e_idx;
        const int* dst = e_idx + E;
        int nb = (E * 4 + 255) / 256;
        hg_edge_logits<<<nb, 256, 0, stream>>>(src, dst, aS, aD, ebuf, mseg, E);
        hg_fix_m<<<(nd * 4 + 255) / 256, 256, 0, stream>>>(mseg, nd * 4);
        hg_edge_exp<<<nb, 256, 0, stream>>>(dst, ebuf, mseg, den, E);
        hg_edge_scatter<<<E, 256, 0, stream>>>(src, dst, hA, ebuf, den, agg);
    };

    float *xdU = bufU[0], *altU = bufU[1];
    float *xdM = bufM[0], *altM = bufM[1];
    float *xdR = bufR[0], *altR = bufR[1];

    for (int L = 0; L < 2; ++L) {
        // ---- xg: lin1(leaky(x)) on first layer, identity afterwards ----
        const float *xu, *xm, *xr;
        if (L == 0) {
            gemm(x_user,  nullptr, nullptr, lin1[0].W, lin1[0].b, nullptr, nullptr, 1.f, 1, 0, 0.f, xgU, nU, HG_D, HG_D, HG_D);
            gemm(x_movie, nullptr, nullptr, lin1[1].W, lin1[1].b, nullptr, nullptr, 1.f, 1, 0, 0.f, xgM, nM, HG_D, HG_D, HG_D);
            gemm(x_review,nullptr, nullptr, lin1[2].W, lin1[2].b, nullptr, nullptr, 1.f, 1, 0, 0.f, xgR, nR, HG_D, HG_D, HG_D);
            xu = xgU; xm = xgM; xr = xgR;
        } else { xu = xdU; xm = xdM; xr = xdR; }

        // ---- four GATConvs (cbr+post accumulate into aggR; mean via 0.5 pre-scale) ----
        gatconv(xm, nM, xr, nR, ei[0], Ecnt[0], gat[L][0], aggR, true,  false); // movie->review
        gatconv(xu, nU, xr, nR, ei[1], Ecnt[1], gat[L][1], aggR, false, false); // user->review
        gatconv(xr, nR, xu, nU, ei[2], Ecnt[2], gat[L][2], aggU, true,  false); // review->user
        gatconv(xm, nM, xm, nM, ei[3], Ecnt[3], gat[L][3], aggM, true,  true ); // movie->movie

        // ---- lin2(leaky(agg + gat bias [0.5x for review])) ----
        gemm(aggU, nullptr, nullptr, lin2[L][0].W, lin2[L][0].b, gat[L][2].b, nullptr, 1.0f, 1, 0, 0.f, altU, nU, HG_D, HG_D, HG_D);
        gemm(aggM, nullptr, nullptr, lin2[L][1].W, lin2[L][1].b, gat[L][3].b, nullptr, 1.0f, 1, 0, 0.f, altM, nM, HG_D, HG_D, HG_D);
        gemm(aggR, nullptr, nullptr, lin2[L][2].W, lin2[L][2].b, gat[L][0].b, gat[L][1].b, 0.5f, 1, 0, 0.f, altR, nR, HG_D, HG_D, HG_D);
        std::swap(xdU, altU); std::swap(xdM, altM); std::swap(xdR, altR);

        // ---- genre mean pooling over concat(review, movie) ----
        fill(seg, 0.f, HG_G * HG_D);
        fill(cntb, 0.f, 64);
        hg_genre_scatter<<<nR, 256, 0, stream>>>(xdR, review_genre, seg, cntb);
        hg_genre_scatter<<<nM, 256, 0, stream>>>(xdM, movie_genre, seg, cntb);
        hg_genre_mean<<<(HG_G * HG_D + 255) / 256, 256, 0, stream>>>(seg, cntb, gA);

        // ---- 2x transformer encoder layers on [30,1024] ----
        float *gcur = gA, *gnx = gB;
        for (int l = 0; l < 2; ++l) {
            const HgTrm& tp = trm[L][l];
            gemm(gcur, nullptr, nullptr, tp.Wqkv, tp.bqkv, nullptr, nullptr, 1.f, 0, 0, 0.f, tqkv, HG_G, HG_D, 3 * HG_D, HG_D);
            hg_mha<<<4, 256, 0, stream>>>(tqkv, cntb, tattn, HG_G);
            gemm(tattn, nullptr, nullptr, tp.Wo, tp.bo, nullptr, nullptr, 1.f, 0, 0, 0.f, ttmp, HG_G, HG_D, HG_D, HG_D);
            hg_ln<<<HG_G, 256, 0, stream>>>(gcur, ttmp, tp.ln1s, tp.ln1b, tx1, HG_D);
            gemm(tx1, nullptr, nullptr, tp.W1, tp.b1, nullptr, nullptr, 1.f, 0, 1, 0.0f, tffn, HG_G, HG_D, 2048, HG_D); // relu
            gemm(tffn, nullptr, nullptr, tp.W2, tp.b2, nullptr, nullptr, 1.f, 0, 0, 0.f, ttmp, HG_G, 2048, HG_D, 2048);
            hg_ln<<<HG_G, 256, 0, stream>>>(tx1, ttmp, tp.ln2s, tp.ln2b, gnx, HG_D);
            std::swap(gcur, gnx);
        }

        // ---- genre-concat linears with gather-A (K=2048, K1=1024), post leaky ----
        gemm(xdR, gcur, review_genre, rlin[L].W, rlin[L].b, nullptr, nullptr, 1.f, 0, 1, 0.01f, altR, nR, 2 * HG_D, HG_D, HG_D);
        std::swap(xdR, altR);
        gemm(xdM, gcur, movie_genre, mlin[L].W, mlin[L].b, nullptr, nullptr, 1.f, 0, 1, 0.01f, altM, nM, 2 * HG_D, HG_D, HG_D);
        std::swap(xdM, altM);
    }

    // ---- outputs: (user, movie, review) concatenated ----
    float* out = (float*)d_out;
    hipMemcpyAsync(out, xdU, (size_t)nU * HG_D * sizeof(float), hipMemcpyDeviceToDevice, stream);
    hipMemcpyAsync(out + (size_t)nU * HG_D, xdM, (size_t)nM * HG_D * sizeof(float), hipMemcpyDeviceToDevice, stream);
    hipMemcpyAsync(out + (size_t)(nU + nM) * HG_D, xdR, (size_t)nR * HG_D * sizeof(float), hipMemcpyDeviceToDevice, stream);
}
